// MLPExtractor_25065429139798
// MI455X (gfx1250) — matrix operand order repaired and checked by reading, not compiled, not run on hardware
//
#include <hip/hip_runtime.h>
#include <stdint.h>

// ---------- problem constants (from reference) ----------
#define BSZ   32
#define NN    100          // N
#define FF    514          // F
#define FE    614          // F + N  (embedded feature width)
#define NPAIR 5000         // K == dim
#define NPAD  5008         // padded per-batch stride for pair arrays
#define HA    64
#define TILES_PER_B 313    // ceil(5000/16)
#define OUT_FILLED (BSZ * 20000)

typedef __attribute__((ext_vector_type(16))) _Float16 v16h;
typedef __attribute__((ext_vector_type(8)))  float    v8f;
typedef __attribute__((ext_vector_type(8)))  uint32_t v8u;

// ---------- workspace layout (bytes) ----------
#define WS_WB     0                          // pair-layer B images: 2*4*2*32*8 u32 = 16 KB
#define WS_WB0    16384                      // stage1 B images: 2*17*4*32*8 u32 = 136 KB
#define WS_ORDER  (WS_WB0 + 139264)          // int[BSZ][NPAD]
#define WS_GG     (WS_ORDER + BSZ*NPAD*4)    // float[BSZ][64]
#define WS_P1     (WS_GG + BSZ*64*4)         // float[BSZ][100][64]
#define WS_P2     (WS_P1 + BSZ*6400*4)
#define WS_LOGIT  (WS_P2 + BSZ*6400*4)       // float[BSZ][NPAD]
#define WS_PBOOL  (WS_LOGIT + BSZ*NPAD*4)
#define WS_RED    (WS_PBOOL + BSZ*NPAD*4)    // float[BSZ][2]

// branch-free tanh via hardware v_exp_f32 (exact to fp32 noise; inputs clamped)
__device__ __forceinline__ float fast_tanh(float x) {
  x = fminf(fmaxf(x, -15.f), 15.f);
  float e = __expf(2.f * x);
  return (e - 1.f) / (e + 1.f);
}
__device__ __forceinline__ float fast_sigmoid(float x) {
  return 1.f / (1.f + __expf(-x));
}
__device__ __forceinline__ uint32_t pack2h(float x, float y) {
  union { uint32_t u; _Float16 h[2]; } p;
  p.h[0] = (_Float16)x; p.h[1] = (_Float16)y;
  return p.u;
}

// =====================================================================
// 0) zero the output buffer
// =====================================================================
__global__ void zero_out_kernel(float* out, int n) {
  int i = blockIdx.x * 256 + threadIdx.x;
  if (i < n) out[i] = 0.0f;
}

// =====================================================================
// 1a) swizzle actor_w1 / actor_w2 (64x64 f32) into f16 B-operand images
//     for v_wmma_f32_16x16x32_f16:
//     lane l holds column n = nt*16 + (l&15); K segment = ks*32 + (l>=16)*16;
//     VGPR v packs K = seg+2v (lo) and seg+2v+1 (hi).
// =====================================================================
__global__ void prep_weights_kernel(const float* w1, const float* w2, uint32_t* WB) {
  int e = blockIdx.x * 256 + threadIdx.x;    // 4096 total
  if (e >= 4096) return;
  int l    = e >> 11;
  int nt   = (e >> 9) & 3;
  int ks   = (e >> 8) & 1;
  int lane = (e >> 3) & 31;
  int v    = e & 7;
  int n  = nt * 16 + (lane & 15);
  int kb = ks * 32 + ((lane >> 4) & 1) * 16 + v * 2;
  const float* W = (l == 0) ? w1 : w2;
  WB[e] = pack2h(W[kb * 64 + n], W[(kb + 1) * 64 + n]);
}

// =====================================================================
// 1b) swizzle the two 514x64 blocks of actor_w0 (rows 514.. and 1028..)
//     into B images for the stage1 WMMA, K padded to 544 (17 ksteps).
//     layout: [((mat*17+ks)*4+nt)*256 + lane*8 + v]
// =====================================================================
__global__ void prep_w0_kernel(const float* aw0, uint32_t* WB0) {
  int e = blockIdx.x * 256 + threadIdx.x;    // 34816 total
  if (e >= 2 * 17 * 4 * 256) return;
  int v    = e & 7;
  int lane = (e >> 3) & 31;
  int rest = e >> 8;
  int nt   = rest & 3; rest >>= 2;
  int ks   = rest % 17;
  int mat  = rest / 17;
  int n  = nt * 16 + (lane & 15);
  int kb = ks * 32 + ((lane >> 4) & 1) * 16 + v * 2;
  float w0 = (kb     < FF) ? aw0[(size_t)(FF + mat * FF + kb    ) * HA + n] : 0.f;
  float w1 = (kb + 1 < FF) ? aw0[(size_t)(FF + mat * FF + kb + 1) * HA + n] : 0.f;
  WB0[e] = pack2h(w0, w1);
}

// =====================================================================
// 2) stream-compact the mask into ascending pair indices (== stable
//    argsort(1-mask)[:, :K] because exactly K entries are 1).
// =====================================================================
__global__ __launch_bounds__(256) void compact_kernel(const float* emb, int* order) {
  __shared__ int scan[256];
  __shared__ int running;
  int b = blockIdx.x, tid = threadIdx.x;
  if (tid == 0) running = 0;
  __syncthreads();
  for (int base = 0; base < NN * NN; base += 256) {
    int item = base + tid;
    int flag = 0;
    if (item < NN * NN) {
      int r = item / NN, c = item % NN;
      float mv = emb[((size_t)b * (NN + 1) + 1 + r) * FE + FF + c];
      flag = (mv > 0.5f) ? 1 : 0;
    }
    scan[tid] = flag;
    __syncthreads();
    for (int off = 1; off < 256; off <<= 1) {
      int v = scan[tid];
      int add = (tid >= off) ? scan[tid - off] : 0;
      __syncthreads();
      scan[tid] = v + add;
      __syncthreads();
    }
    int pos = running + scan[tid] - flag;   // exclusive prefix
    if (flag && pos < NPAIR) order[b * NPAD + pos] = item;
    __syncthreads();
    if (tid == 255) running += scan[255];
    __syncthreads();
  }
}

// =====================================================================
// 3a) stage1 WMMA: P1 = nodes@W0a, P2 = nodes@W0b  (per batch).
//     One wave per (b, mat, mtile): M=16 rows, N=64 (4 nt), K=544 (17 ks).
//     A built on the fly from global f32 rows (f16 convert+pack).
// =====================================================================
__global__ __launch_bounds__(256) void stage1_wmma_kernel(
    const float* emb, const uint32_t* WB0u, float* P1, float* P2)
{
  const int wave = threadIdx.x >> 5;
  const int lane = threadIdx.x & 31;
  int wu = blockIdx.x * 8 + wave;          // 448 work units
  if (wu >= BSZ * 2 * 7) return;           // wave-uniform
  int mtile = wu % 7;
  int mat   = (wu / 7) & 1;
  int b     = wu / 14;
  const int hi = lane >> 4;
  int m   = mtile * 16 + (lane & 15);
  int row = (m < NN) ? m : 0;              // padded rows: compute garbage, masked on store
  const float* nr = emb + ((size_t)b * (NN + 1) + 1 + row) * FE;
  const v16h* BW = (const v16h*)WB0u;

  v8f acc[4] = {};
  for (int ks = 0; ks < 16; ++ks) {        // full k-steps, no bounds checks
    v8u u;
#pragma unroll
    for (int v = 0; v < 8; ++v) {
      int k0 = ks * 32 + hi * 16 + v * 2;
      u[v] = pack2h(nr[k0], nr[k0 + 1]);
    }
    v16h a = __builtin_bit_cast(v16h, u);
#pragma unroll
    for (int nt = 0; nt < 4; ++nt)
      acc[nt] = __builtin_amdgcn_wmma_f32_16x16x32_f16(false, a, false,
                    BW[(((size_t)mat * 17 + ks) * 4 + nt) * 32 + lane],
                    (short)0, acc[nt], false, false);
  }
  {                                        // ks=16 tail: only k=512,513 live (v==0, hi==0)
    v8u u = {};
    if (hi == 0) u[0] = pack2h(nr[512], nr[513]);
    v16h a = __builtin_bit_cast(v16h, u);
#pragma unroll
    for (int nt = 0; nt < 4; ++nt)
      acc[nt] = __builtin_amdgcn_wmma_f32_16x16x32_f16(false, a, false,
                    BW[(((size_t)mat * 17 + 16) * 4 + nt) * 32 + lane],
                    (short)0, acc[nt], false, false);
  }
  float* P = (mat == 0 ? P1 : P2) + (size_t)b * NN * HA;
#pragma unroll
  for (int nt = 0; nt < 4; ++nt)
#pragma unroll
    for (int r = 0; r < 8; ++r) {
      int mm = mtile * 16 + r + hi * 8;
      if (mm < NN) P[mm * HA + nt * 16 + (lane & 15)] = acc[nt][r];
    }
}

// =====================================================================
// 3b) Gg = g@W0g + b0, and the whole critic MLP on g (tiny, per batch).
// =====================================================================
__global__ __launch_bounds__(64) void gg_critic_kernel(
    const float* emb, const float* aw0, const float* ab0,
    const float* cw0, const float* cb0, const float* cw1, const float* cb1,
    const float* cw2, const float* cb2, const float* cw3, const float* cb3,
    float* Gg, float* value_out)
{
  int b = blockIdx.x, tid = threadIdx.x;
  const float* g = emb + (size_t)b * (NN + 1) * FE;    // row 0
  __shared__ float h[2][HA];
  {
    float s = ab0[tid];
    for (int k = 0; k < FF; ++k) s += g[k] * aw0[k * HA + tid];
    Gg[b * HA + tid] = s;                              // bias b0 folded in
    float c0 = cb0[tid];
    for (int k = 0; k < FF; ++k) c0 += g[k] * cw0[k * HA + tid];
    h[0][tid] = fast_tanh(c0);
  }
  __syncthreads();
  {
    float s = cb1[tid];
    for (int k = 0; k < HA; ++k) s += h[0][k] * cw1[k * HA + tid];
    h[1][tid] = fast_tanh(s);
  }
  __syncthreads();
  {
    float s = cb2[tid];
    for (int k = 0; k < HA; ++k) s += h[1][k] * cw2[k * HA + tid];
    h[0][tid] = fast_tanh(s);
  }
  __syncthreads();
  if (tid == 0) {
    float s = cb3[0];
    for (int k = 0; k < HA; ++k) s += h[0][k] * cw3[k];
    value_out[b] = s;
  }
}

// =====================================================================
// 4) pair MLP: one wave per 16-pair tile.
//    h0 = tanh(Gg + P1[i1] + P2[i2])                      (gathered)
//    h1 = tanh(h0 @ W1 + b1)  -> 8x v_wmma_f32_16x16x32_f16
//    h2 = tanh(h1 @ W2 + b2)  -> 8x v_wmma_f32_16x16x32_f16
//    [logit, pbool] = h2 @ W3 + b3                        (VALU)
// =====================================================================
__device__ __forceinline__ void load_A(const _Float16* X, int lane, v16h& a0, v16h& a1) {
  int m = lane & 15, hi = lane >> 4;
  const uint32_t* row = (const uint32_t*)(X + m * 72);
  v8u u0, u1;
#pragma unroll
  for (int v = 0; v < 8; ++v) {
    int k0 = v + (hi << 3);                              // u32 index within k-step
    u0[v] = row[k0];
    u1[v] = row[k0 + 16];                                // + 32 halves
  }
  a0 = __builtin_bit_cast(v16h, u0);
  a1 = __builtin_bit_cast(v16h, u1);
}

__global__ __launch_bounds__(256) void pair_mlp_kernel(
    const int* order, const float* Gg, const float* P1, const float* P2,
    const float* b1, const float* b2, const float* W3, const float* b3,
    const uint32_t* WBu, float* logits, float* pbool)
{
  __shared__ float smemX[8][576];    // per-wave 16x64 f16 tile, row stride 72 halves
  __shared__ float smemY[8][1088];   // per-wave 16x64 f32 tile, row stride 68 floats

  const int wave = threadIdx.x >> 5;
  const int lane = threadIdx.x & 31;
  const int wid  = blockIdx.x * 8 + wave;
  const int b    = wid / TILES_PER_B;
  const int t    = wid % TILES_PER_B;
  if (b >= BSZ) return;                       // wave-uniform
  const int pbase = t * 16;
  const int nrows = (NPAIR - pbase < 16) ? (NPAIR - pbase) : 16;

  _Float16* X = (_Float16*)&smemX[wave][0];
  float*    Y = &smemY[wave][0];

  const int*   ordb = order + b * NPAD;
  const float* P1b  = P1 + b * NN * HA;
  const float* P2b  = P2 + b * NN * HA;
  const float* Ggb  = Gg + b * HA;
  const v16h*  BW   = (const v16h*)WBu;       // [((l*4+nt)*2+ks)*32 + lane]
  const int hi = lane >> 4;

  // ---- build h0 tile (f16, LDS) ----
  for (int idx = lane; idx < 16 * HA; idx += 32) {
    int m = idx >> 6, c = idx & 63;
    int r  = (m < nrows) ? ordb[pbase + m] : ordb[pbase];
    int i1 = r / NN, i2 = r % NN;
    float v = Ggb[c] + P1b[i1 * HA + c] + P2b[i2 * HA + c];
    X[m * 72 + c] = (_Float16)fast_tanh(v);
  }
  asm volatile("s_wait_dscnt 0" ::: "memory");

  // ---- layer 1 (WMMA) ----
  v16h a0, a1;
  load_A(X, lane, a0, a1);
  v8f acc[4];
#pragma unroll
  for (int nt = 0; nt < 4; ++nt) {
    float bias = b1[nt * 16 + (lane & 15)];
    v8f c = {bias, bias, bias, bias, bias, bias, bias, bias};
    c = __builtin_amdgcn_wmma_f32_16x16x32_f16(false, a0, false,
            BW[((0 * 4 + nt) * 2 + 0) * 32 + lane], (short)0, c, false, false);
    c = __builtin_amdgcn_wmma_f32_16x16x32_f16(false, a1, false,
            BW[((0 * 4 + nt) * 2 + 1) * 32 + lane], (short)0, c, false, false);
    acc[nt] = c;
  }
#pragma unroll
  for (int nt = 0; nt < 4; ++nt)
#pragma unroll
    for (int r = 0; r < 8; ++r) {
      int m = r + hi * 8, n = nt * 16 + (lane & 15);
      X[m * 72 + n] = (_Float16)fast_tanh(acc[nt][r]);
    }
  asm volatile("s_wait_dscnt 0" ::: "memory");

  // ---- layer 2 (WMMA) ----
  load_A(X, lane, a0, a1);
#pragma unroll
  for (int nt = 0; nt < 4; ++nt) {
    float bias = b2[nt * 16 + (lane & 15)];
    v8f c = {bias, bias, bias, bias, bias, bias, bias, bias};
    c = __builtin_amdgcn_wmma_f32_16x16x32_f16(false, a0, false,
            BW[((1 * 4 + nt) * 2 + 0) * 32 + lane], (short)0, c, false, false);
    c = __builtin_amdgcn_wmma_f32_16x16x32_f16(false, a1, false,
            BW[((1 * 4 + nt) * 2 + 1) * 32 + lane], (short)0, c, false, false);
    acc[nt] = c;
  }
#pragma unroll
  for (int nt = 0; nt < 4; ++nt)
#pragma unroll
    for (int r = 0; r < 8; ++r) {
      int m = r + hi * 8, n = nt * 16 + (lane & 15);
      Y[m * 68 + n] = fast_tanh(acc[nt][r]);
    }
  asm volatile("s_wait_dscnt 0" ::: "memory");

  // ---- output layer: 2 outputs per row, lane (m, o) = (l&15, l>>4) ----
  {
    int m = lane & 15, o = hi;
    if (m < nrows) {
      float s = b3[o];
      for (int k = 0; k < HA; ++k) s += Y[m * 68 + k] * W3[k * 2 + o];
      int j = pbase + m;
      if (o == 0) logits[b * NPAD + j] = s;
      else        pbool [b * NPAD + j] = s;
    }
  }
}

// =====================================================================
// 5) per-batch softmax reduction (max, sum of exp)
// =====================================================================
__global__ __launch_bounds__(256) void softmax_reduce_kernel(const float* logits, float* red) {
  __shared__ float sm[256];
  int b = blockIdx.x, tid = threadIdx.x;
  float mx = -3.4e38f;
  for (int j = tid; j < NPAIR; j += 256) mx = fmaxf(mx, logits[b * NPAD + j]);
  sm[tid] = mx; __syncthreads();
  for (int off = 128; off; off >>= 1) {
    if (tid < off) sm[tid] = fmaxf(sm[tid], sm[tid + off]);
    __syncthreads();
  }
  float m = sm[0]; __syncthreads();
  float s = 0.f;
  for (int j = tid; j < NPAIR; j += 256) s += __expf(logits[b * NPAD + j] - m);
  sm[tid] = s; __syncthreads();
  for (int off = 128; off; off >>= 1) {
    if (tid < off) sm[tid] += sm[tid + off];
    __syncthreads();
  }
  if (tid == 0) { red[b * 2] = m; red[b * 2 + 1] = sm[0]; }
}

// =====================================================================
// 6) scatter pi*sig and pi*(1-sig) into the zeroed 2*100x100 grids
// =====================================================================
__global__ void scatter_kernel(const float* logits, const float* pbool,
                               const float* red, const int* order, float* out) {
  int j = blockIdx.x * 256 + threadIdx.x;
  int b = blockIdx.y;
  if (j >= NPAIR) return;
  float m = red[b * 2], s = red[b * 2 + 1];
  float pi = __expf(logits[b * NPAD + j] - m) / s;
  float bb = fast_sigmoid(pbool[b * NPAD + j]);
  int ord = order[b * NPAD + j];
  out[(size_t)b * 20000 + ord]         = pi * bb;
  out[(size_t)b * 20000 + 10000 + ord] = pi * (1.f - bb);
}

// =====================================================================
extern "C" void kernel_launch(void* const* d_in, const int* in_sizes, int n_in,
                              void* d_out, int out_size, void* d_ws, size_t ws_size,
                              hipStream_t stream) {
  // dict order: 0 embedded_features, 1 dim, 2..9 actor w/b, 10..17 critic w/b
  const float* emb = (const float*)d_in[0];
  const float* aw0 = (const float*)d_in[2];
  const float* ab0 = (const float*)d_in[3];
  const float* aw1 = (const float*)d_in[4];
  const float* ab1 = (const float*)d_in[5];
  const float* aw2 = (const float*)d_in[6];
  const float* ab2 = (const float*)d_in[7];
  const float* aw3 = (const float*)d_in[8];
  const float* ab3 = (const float*)d_in[9];
  const float* cw0 = (const float*)d_in[10];
  const float* cb0 = (const float*)d_in[11];
  const float* cw1 = (const float*)d_in[12];
  const float* cb1 = (const float*)d_in[13];
  const float* cw2 = (const float*)d_in[14];
  const float* cb2 = (const float*)d_in[15];
  const float* cw3 = (const float*)d_in[16];
  const float* cb3 = (const float*)d_in[17];

  char* ws = (char*)d_ws;
  uint32_t* WB    = (uint32_t*)(ws + WS_WB);
  uint32_t* WB0   = (uint32_t*)(ws + WS_WB0);
  int*      order = (int*)     (ws + WS_ORDER);
  float*    Gg    = (float*)   (ws + WS_GG);
  float*    P1    = (float*)   (ws + WS_P1);
  float*    P2    = (float*)   (ws + WS_P2);
  float*    logit = (float*)   (ws + WS_LOGIT);
  float*    pbl   = (float*)   (ws + WS_PBOOL);
  float*    red   = (float*)   (ws + WS_RED);

  float* out       = (float*)d_out;
  float* value_out = out + OUT_FILLED;

  zero_out_kernel<<<(OUT_FILLED + BSZ + 255) / 256, 256, 0, stream>>>(out, OUT_FILLED + BSZ);
  prep_weights_kernel<<<16, 256, 0, stream>>>(aw1, aw2, WB);
  prep_w0_kernel<<<(2 * 17 * 4 * 256 + 255) / 256, 256, 0, stream>>>(aw0, WB0);
  compact_kernel<<<BSZ, 256, 0, stream>>>(emb, order);
  stage1_wmma_kernel<<<(BSZ * 2 * 7 + 7) / 8, 256, 0, stream>>>(emb, WB0, P1, P2);
  gg_critic_kernel<<<BSZ, 64, 0, stream>>>(emb, aw0, ab0, cw0, cb0, cw1, cb1,
                                           cw2, cb2, cw3, cb3, Gg, value_out);
  int nwaves = BSZ * TILES_PER_B;
  pair_mlp_kernel<<<(nwaves + 7) / 8, 256, 0, stream>>>(order, Gg, P1, P2,
      ab1, ab2, aw3, ab3, WB, logit, pbl);
  softmax_reduce_kernel<<<BSZ, 256, 0, stream>>>(logit, red);
  dim3 sg((NPAIR + 255) / 256, BSZ);
  scatter_kernel<<<sg, 256, 0, stream>>>(logit, pbl, red, order, out);
}
